// BRASKModel_35845797052821
// MI455X (gfx1250) — matrix operand order repaired and checked by reading, not compiled, not run on hardware
//
#include <hip/hip_runtime.h>
#include <math.h>

constexpr int kB = 8, kL = 128, kH = 768, kR = 32;

typedef __bf16 bf16;
typedef __attribute__((ext_vector_type(8)))  __bf16 v8bf;
typedef __attribute__((ext_vector_type(16))) __bf16 v16bf;
typedef __attribute__((ext_vector_type(8)))  float  v8f;

// ---------------------------------------------------------------------------
// WMMA GEMM: C[M,N] = A[M,K](bf16) * W[K,N] + bias,  W pre-transposed as Wt[N][K]
// One wave -> 16(M) x 64(N) tile, K-loop step 32 using v_wmma_f32_16x16x32_bf16.
// Requires M%16==0, N%64==0, K%32==0, EXEC all ones (uniform control flow).
// Batched via blockIdx.y with element strides.
// ---------------------------------------------------------------------------
union AFrag { v16bf v; v8bf h[2]; };

__global__ void wmma_gemm(const bf16* __restrict__ A, const bf16* __restrict__ Wt,
                          const float* __restrict__ bias, float* __restrict__ C,
                          int M, int N, int K,
                          long sA, long sW, long sC) {
  const int lane = threadIdx.x & 31;
  const int wid  = blockIdx.x * (blockDim.x >> 5) + (threadIdx.x >> 5);
  const int ntn  = N >> 6;
  const int tm = wid / ntn, tn = wid % ntn;
  if (tm >= (M >> 4)) return;                 // wave-uniform exit
  A  += (long)blockIdx.y * sA;
  Wt += (long)blockIdx.y * sW;
  C  += (long)blockIdx.y * sC;

  const int half = lane >> 4;
  const int l15  = lane & 15;
  const int m  = (tm << 4) + l15;
  const int n0 = tn << 6;

  const bf16* arow = A + (long)m * K;
  const bf16* bp0 = Wt + (long)(n0 + l15)      * K + half * 16;
  const bf16* bp1 = Wt + (long)(n0 + 16 + l15) * K + half * 16;
  const bf16* bp2 = Wt + (long)(n0 + 32 + l15) * K + half * 16;
  const bf16* bp3 = Wt + (long)(n0 + 48 + l15) * K + half * 16;

  v8f acc0 = {}, acc1 = {}, acc2 = {}, acc3 = {};
  for (int kk = 0; kk < K; kk += 32) {
    AFrag a;
    a.h[0] = *(const v8bf*)(arow + kk + half * 16);       // K = h*16 .. h*16+7
    a.h[1] = *(const v8bf*)(arow + kk + half * 16 + 8);   // K = h*16+8 .. h*16+15
    v16bf b0 = *(const v16bf*)(bp0 + kk);                 // K = h*16 .. h*16+15
    v16bf b1 = *(const v16bf*)(bp1 + kk);
    v16bf b2 = *(const v16bf*)(bp2 + kk);
    v16bf b3 = *(const v16bf*)(bp3 + kk);
    acc0 = __builtin_amdgcn_wmma_f32_16x16x32_bf16(false, a.v, false, b0, (short)0, acc0, false, false);
    acc1 = __builtin_amdgcn_wmma_f32_16x16x32_bf16(false, a.v, false, b1, (short)0, acc1, false, false);
    acc2 = __builtin_amdgcn_wmma_f32_16x16x32_bf16(false, a.v, false, b2, (short)0, acc2, false, false);
    acc3 = __builtin_amdgcn_wmma_f32_16x16x32_bf16(false, a.v, false, b3, (short)0, acc3, false, false);
  }

  const int row0 = (tm << 4) + half * 8;
  float bs0 = bias ? bias[n0 + l15]      : 0.f;
  float bs1 = bias ? bias[n0 + 16 + l15] : 0.f;
  float bs2 = bias ? bias[n0 + 32 + l15] : 0.f;
  float bs3 = bias ? bias[n0 + 48 + l15] : 0.f;
  #pragma unroll
  for (int j = 0; j < 8; ++j) {
    long r = (long)(row0 + j) * N;
    C[r + n0 + l15]      = acc0[j] + bs0;
    C[r + n0 + 16 + l15] = acc1[j] + bs1;
    C[r + n0 + 32 + l15] = acc2[j] + bs2;
    C[r + n0 + 48 + l15] = acc3[j] + bs3;
  }
}

// ---------------------------------------------------------------------------
// f32 -> bf16 with 2-D padding (also plain convert when Rp==R, Cp==C)
// ---------------------------------------------------------------------------
__global__ void k_pad_bf16(const float* __restrict__ src, bf16* __restrict__ dst,
                           int R, int C, int Rp, int Cp) {
  int i = blockIdx.x * blockDim.x + threadIdx.x;
  if (i >= Rp * Cp) return;
  int r = i / Cp, c = i % Cp;
  dst[i] = (r < R && c < C) ? (bf16)src[(long)r * C + c] : (bf16)0.f;
}

// src [R][C] f32 -> dst [C][Rp] bf16 (transpose + K-pad), batched
__global__ void k_transpose_bf16(const float* __restrict__ src, bf16* __restrict__ dst,
                                 int R, int C, int Rp, long sb, long db) {
  src += (long)blockIdx.y * sb;
  dst += (long)blockIdx.y * db;
  int i = blockIdx.x * blockDim.x + threadIdx.x;
  if (i >= C * Rp) return;
  int c = i / Rp, r = i % Rp;
  dst[i] = (r < R) ? (bf16)src[(long)r * C + c] : (bf16)0.f;
}

__device__ inline float wave_sum(float x) {
  #pragma unroll
  for (int o = 16; o > 0; o >>= 1) x += __shfl_down(x, o, 32);
  return x;
}
__device__ inline float sigmoidf(float x) { return 1.f / (1.f + __expf(-x)); }

// f_ssp/f_sep: one wave per (b,l) row
__global__ void k_sub_probs(const float* __restrict__ tok,
                            const float* ws, const float* wsb,
                            const float* we, const float* web,
                            float* __restrict__ ssp, float* __restrict__ sep) {
  int row  = blockIdx.x * (blockDim.x >> 5) + (threadIdx.x >> 5);
  int lane = threadIdx.x & 31;
  if (row >= kB * kL) return;
  const float* t = tok + (long)row * kH;
  float s0 = 0.f, s1 = 0.f;
  for (int h = lane; h < kH; h += 32) { float x = t[h]; s0 += x * ws[h]; s1 += x * we[h]; }
  s0 = wave_sum(s0); s1 = wave_sum(s1);
  if (lane == 0) { ssp[row] = sigmoidf(s0 + wsb[0]); sep[row] = sigmoidf(s1 + web[0]); }
}

// exact two-pointer span decode, one thread per batch element
__global__ void k_extract(const float* __restrict__ ssp, const float* __restrict__ sep,
                          int* __restrict__ sidx, int* __restrict__ eidx,
                          float* __restrict__ maskf) {
  int b = blockIdx.x * blockDim.x + threadIdx.x;
  if (b >= kB) return;
  int starts[kL], ends[kL];
  int ns = 0, ne = 0;
  for (int i = 0; i < kL; ++i) if (ssp[b * kL + i] > 0.5f) starts[ns++] = i;
  for (int i = 0; i < kL; ++i) if (sep[b * kL + i] > 0.5f) ends[ne++] = i;
  int p = 0;
  for (int slot = 0; slot < kL; ++slot) {
    int s = (slot < ns) ? starts[slot] : kL;
    int ss = 0;
    while (ss < ne && ends[ss] < s) ++ss;     // searchsorted (pads value kL >= s)
    int p2 = p > ss ? p : ss;
    bool matched = (s < kL) && (p2 < ne);
    sidx[b * kL + slot]  = matched ? s : 0;
    eidx[b * kL + slot]  = matched ? ends[p2] : 0;
    maskf[b * kL + slot] = matched ? 1.f : 0.f;
    p = matched ? p2 + 1 : p2;
  }
}

// f_pad bf16: 0.5*(tok[s]+tok[e]) masked
__global__ void k_build_pad(const float* __restrict__ tok, const int* __restrict__ sidx,
                            const int* __restrict__ eidx, const float* __restrict__ maskf,
                            bf16* __restrict__ padbf) {
  int i = blockIdx.x * blockDim.x + threadIdx.x;
  if (i >= kB * kL * kH) return;
  int m = i / kH, h = i % kH;
  int b = m / kL;
  float v = 0.f;
  if (maskf[m] != 0.f) {
    v = 0.5f * (tok[((long)b * kL + sidx[m]) * kH + h] +
                tok[((long)b * kL + eidx[m]) * kH + h]);
  }
  padbf[i] = (bf16)v;
}

// v[b][r][l] = sum_h tanh(xw[b,l,h] + gw[b,h] + rw[r,h]) * Vw[h] + Vb
__global__ void k_attn(const float* __restrict__ xw, const float* __restrict__ gw,
                       const float* __restrict__ rw, const float* __restrict__ Vw,
                       const float* __restrict__ Vb, float* __restrict__ v) {
  int bl = blockIdx.x;             // (b*kL + l)
  int b = bl / kL, l = bl % kL;
  __shared__ float base[kH];
  for (int h = threadIdx.x; h < kH; h += blockDim.x)
    base[h] = xw[(long)bl * kH + h] + gw[(long)b * kH + h];
  __syncthreads();
  int wv = threadIdx.x >> 5, lane = threadIdx.x & 31;
  for (int r = wv; r < kR; r += 8) {
    float s = 0.f;
    for (int h = lane; h < kH; h += 32)
      s += tanhf(base[h] + rw[(long)r * kH + h]) * Vw[h];
    s = wave_sum(s);
    if (lane == 0) v[((long)b * kR + r) * kL + l] = s + Vb[0];
  }
}

// softmax over L per (b,r) row, store bf16 attention weights
__global__ void k_softmax(const float* __restrict__ v, bf16* __restrict__ abf) {
  int row  = blockIdx.x * (blockDim.x >> 5) + (threadIdx.x >> 5);
  int lane = threadIdx.x & 31;
  if (row >= kB * kR) return;
  const float* p = v + (long)row * kL;
  float x[4], mx = -3.4e38f;
  #pragma unroll
  for (int i = 0; i < 4; ++i) { x[i] = p[lane + 32 * i]; mx = fmaxf(mx, x[i]); }
  #pragma unroll
  for (int o = 16; o > 0; o >>= 1) mx = fmaxf(mx, __shfl_xor(mx, o, 32));
  float sum = 0.f;
  #pragma unroll
  for (int i = 0; i < 4; ++i) { x[i] = __expf(x[i] - mx); sum += x[i]; }
  #pragma unroll
  for (int o = 16; o > 0; o >>= 1) sum += __shfl_xor(sum, o, 32);
  float inv = 1.f / sum;
  #pragma unroll
  for (int i = 0; i < 4; ++i) abf[(long)row * kL + lane + 32 * i] = (bf16)(x[i] * inv);
}

// s2[p][b*R+r] = dot(fC/bC row, head weight)
__global__ void k_s2(const float* __restrict__ fC, const float* __restrict__ bC,
                     const float* wfso, const float* wfeo,
                     const float* wbss, const float* wbse,
                     float* __restrict__ s2) {
  int row  = blockIdx.x * (blockDim.x >> 5) + (threadIdx.x >> 5);
  int lane = threadIdx.x & 31;
  if (row >= kB * kR) return;
  const float* fc = fC + (long)row * kH;
  const float* bc = bC + (long)row * kH;
  float a = 0.f, bsum = 0.f, c = 0.f, d = 0.f;
  for (int h = lane; h < kH; h += 32) {
    float f = fc[h], g = bc[h];
    a += f * wfso[h]; bsum += f * wfeo[h]; c += g * wbss[h]; d += g * wbse[h];
  }
  a = wave_sum(a); bsum = wave_sum(bsum); c = wave_sum(c); d = wave_sum(d);
  if (lane == 0) {
    s2[0 * kB * kR + row] = a;    s2[1 * kB * kR + row] = bsum;
    s2[2 * kB * kR + row] = c;    s2[3 * kB * kR + row] = d;
  }
}

// s1[p][b*L+l] = dot(mask*W_s-out + Wx2-out + tok, head weight)
__global__ void k_s1(const float* __restrict__ tok,
                     const float* __restrict__ fsw, const float* __restrict__ fx2,
                     const float* __restrict__ bow, const float* __restrict__ bx2,
                     const float* __restrict__ maskf,
                     const float* wfso, const float* wfeo,
                     const float* wbss, const float* wbse,
                     float* __restrict__ s1) {
  int row  = blockIdx.x * (blockDim.x >> 5) + (threadIdx.x >> 5);
  int lane = threadIdx.x & 31;
  if (row >= kB * kL) return;
  float mk = maskf[row];
  long base = (long)row * kH;
  float a = 0.f, bsum = 0.f, c = 0.f, d = 0.f;
  for (int h = lane; h < kH; h += 32) {
    float t  = tok[base + h];
    float Hf = mk * fsw[base + h] + fx2[base + h] + t;
    float Hb = mk * bow[base + h] + bx2[base + h] + t;
    a += Hf * wfso[h]; bsum += Hf * wfeo[h]; c += Hb * wbss[h]; d += Hb * wbse[h];
  }
  a = wave_sum(a); bsum = wave_sum(bsum); c = wave_sum(c); d = wave_sum(d);
  if (lane == 0) {
    s1[0 * kB * kL + row] = a;    s1[1 * kB * kL + row] = bsum;
    s1[2 * kB * kL + row] = c;    s1[3 * kB * kL + row] = d;
  }
}

__global__ void k_final(const float* __restrict__ s1, const float* __restrict__ s2,
                        const float* bfso, const float* bfeo,
                        const float* bbss, const float* bbse,
                        float* __restrict__ out) {
  int i = blockIdx.x * blockDim.x + threadIdx.x;
  const int tot = 4 * kB * kL * kR;
  if (i >= tot) return;
  int p = i / (kB * kL * kR);
  int rem = i % (kB * kL * kR);
  int b = rem / (kL * kR);
  int lr = rem % (kL * kR);
  int l = lr / kR, r = lr % kR;
  float bias = (p == 0) ? bfso[0] : (p == 1) ? bfeo[0] : (p == 2) ? bbss[0] : bbse[0];
  float x = s1[p * kB * kL + b * kL + l] + s2[p * kB * kR + b * kR + r] + bias;
  out[i] = sigmoidf(x);
}

// ---------------------------------------------------------------------------
static void launch_gemm(const bf16* A, const bf16* Wt, const float* bias, float* C,
                        int M, int N, int K, int batch, long sA, long sW, long sC,
                        hipStream_t st) {
  int waves = (M / 16) * (N / 64);
  int blocks = (waves + 3) / 4;                // 4 waves (128 threads) per block
  wmma_gemm<<<dim3(blocks, batch), 128, 0, st>>>(A, Wt, bias, C, M, N, K, sA, sW, sC);
}

extern "C" void kernel_launch(void* const* d_in, const int* in_sizes, int n_in,
                              void* d_out, int out_size, void* d_ws, size_t ws_size,
                              hipStream_t stream) {
  auto F = [&](int i) { return (const float*)d_in[i]; };
  const float* embs = F(0);          // (8,128,768)
  const float* h_gs = F(1);          // (8,768)
  // head weights/biases (dict insertion order)
  const float *fss_w = F(2),  *fss_b = F(3);
  const float *fes_w = F(4),  *fes_b = F(5);
  const float *fso_w = F(6),  *fso_b = F(7);
  const float *feo_w = F(8),  *feo_b = F(9);
  // b_start_obj (10,11), b_end_obj (12,13) unused by reference outputs
  const float *bss_w = F(14), *bss_b = F(15);
  const float *bse_w = F(16), *bse_b = F(17);
  const float *V_w   = F(18), *V_b   = F(19);
  const float *fWs_w = F(20), *fWs_b = F(21);
  const float *bWs_w = F(22), *bWs_b = F(23);
  const float *fWr_w = F(24), *fWr_b = F(25);
  const float *fWg_w = F(26), *fWg_b = F(27);
  const float *fWx_w = F(28), *fWx_b = F(29);
  const float *bWr_w = F(30), *bWr_b = F(31);
  const float *bWg_w = F(32), *bWg_b = F(33);
  const float *bWx_w = F(34), *bWx_b = F(35);
  const float *fW2_w = F(36), *fW2_b = F(37);
  const float *bW2_w = F(38), *bW2_b = F(39);
  const float *rp_w  = F(40), *rp_b  = F(41);
  const float *rel_e = F(42);        // (32,768)
  const float *rel_t = F(43);        // (32,100)

  // ---- workspace bump allocator (256B aligned) ----
  char* ws = (char*)d_ws;
  size_t off = 0;
  auto alloc = [&](size_t bytes) { void* p = ws + off; off = (off + bytes + 255) & ~(size_t)255; return p; };

  const int ML = kB * kL;            // 1024
  bf16*  Abf   = (bf16*)alloc((size_t)ML * kH * 2);
  bf16*  tokT  = (bf16*)alloc((size_t)kB * kH * kL * 2);
  bf16*  padbf = (bf16*)alloc((size_t)ML * kH * 2);
  bf16*  hgsb  = (bf16*)alloc((size_t)16 * kH * 2);
  bf16*  frelb = (bf16*)alloc((size_t)kR * kH * 2);
  bf16*  rtb   = (bf16*)alloc((size_t)kR * 128 * 2);
  bf16*  brelb = (bf16*)alloc((size_t)kR * kH * 2);
  bf16*  Wt[10];
  const float* Wsrc[10] = {fWs_w, bWs_w, fWr_w, fWg_w, fWx_w, bWr_w, bWg_w, bWx_w, fW2_w, bW2_w};
  for (int i = 0; i < 10; ++i) Wt[i] = (bf16*)alloc((size_t)kH * kH * 2);
  bf16*  rpT   = (bf16*)alloc((size_t)kH * 128 * 2);
  float* fx    = (float*)alloc((size_t)ML * kH * 4);
  float* bx    = (float*)alloc((size_t)ML * kH * 4);
  float* fx2   = (float*)alloc((size_t)ML * kH * 4);
  float* bx2   = (float*)alloc((size_t)ML * kH * 4);
  float* fsw   = (float*)alloc((size_t)ML * kH * 4);
  float* bow   = (float*)alloc((size_t)ML * kH * 4);
  float* fr    = (float*)alloc((size_t)kR * kH * 4);
  float* br    = (float*)alloc((size_t)kR * kH * 4);
  float* brelf = (float*)alloc((size_t)kR * kH * 4);
  float* fg    = (float*)alloc((size_t)16 * kH * 4);
  float* bg    = (float*)alloc((size_t)16 * kH * 4);
  float* fv    = (float*)alloc((size_t)kB * kR * kL * 4);
  float* bv    = (float*)alloc((size_t)kB * kR * kL * 4);
  bf16*  fAb   = (bf16*)alloc((size_t)kB * kR * kL * 2);
  bf16*  bAb   = (bf16*)alloc((size_t)kB * kR * kL * 2);
  float* fC    = (float*)alloc((size_t)kB * kR * kH * 4);
  float* bC    = (float*)alloc((size_t)kB * kR * kH * 4);
  float* ssp   = (float*)alloc((size_t)ML * 4);
  float* sep   = (float*)alloc((size_t)ML * 4);
  int*   sidx  = (int*)alloc((size_t)ML * 4);
  int*   eidx  = (int*)alloc((size_t)ML * 4);
  float* maskf = (float*)alloc((size_t)ML * 4);
  float* s1    = (float*)alloc((size_t)4 * ML * 4);
  float* s2    = (float*)alloc((size_t)4 * kB * kR * 4);
  (void)ws_size; (void)in_sizes; (void)n_in; (void)out_size;

  const int T = 256;
  auto blk = [&](long n) { return (int)((n + T - 1) / T); };

  // ---- conversions / transposes ----
  k_pad_bf16<<<blk((long)ML * kH), T, 0, stream>>>(embs, Abf, ML, kH, ML, kH);
  k_transpose_bf16<<<dim3(blk((long)kH * kL), kB), T, 0, stream>>>(
      embs, tokT, kL, kH, kL, (long)kL * kH, (long)kH * kL);
  k_pad_bf16<<<blk((long)16 * kH), T, 0, stream>>>(h_gs, hgsb, kB, kH, 16, kH);
  k_pad_bf16<<<blk((long)kR * kH), T, 0, stream>>>(rel_e, frelb, kR, kH, kR, kH);
  k_pad_bf16<<<blk((long)kR * 128), T, 0, stream>>>(rel_t, rtb, kR, 100, kR, 128);
  for (int i = 0; i < 10; ++i)
    k_transpose_bf16<<<dim3(blk((long)kH * kH), 1), T, 0, stream>>>(
        Wsrc[i], Wt[i], kH, kH, kH, 0, 0);
  k_transpose_bf16<<<dim3(blk((long)kH * 128), 1), T, 0, stream>>>(
      rp_w, rpT, 100, kH, 128, 0, 0);

  // ---- subject probs + span decode + padded entity matrix ----
  k_sub_probs<<<blk((long)ML * 32), T, 0, stream>>>(embs, fss_w, fss_b, fes_w, fes_b, ssp, sep);
  k_extract<<<1, 32, 0, stream>>>(ssp, sep, sidx, eidx, maskf);
  k_build_pad<<<blk((long)ML * kH), T, 0, stream>>>(embs, sidx, eidx, maskf, padbf);

  // ---- WMMA GEMMs ----
  launch_gemm(Abf,   Wt[4], fWx_b, fx,  ML, kH, kH, 1, 0, 0, 0, stream);  // tok @ f_W_x
  launch_gemm(Abf,   Wt[7], bWx_b, bx,  ML, kH, kH, 1, 0, 0, 0, stream);  // tok @ b_W_x
  launch_gemm(Abf,   Wt[8], fW2_b, fx2, ML, kH, kH, 1, 0, 0, 0, stream);  // tok @ f_Wx2
  launch_gemm(Abf,   Wt[9], bW2_b, bx2, ML, kH, kH, 1, 0, 0, 0, stream);  // tok @ b_Wx2
  launch_gemm(padbf, Wt[0], fWs_b, fsw, ML, kH, kH, 1, 0, 0, 0, stream);  // pad @ f_W_s
  launch_gemm(padbf, Wt[1], bWs_b, bow, ML, kH, kH, 1, 0, 0, 0, stream);  // pad @ b_W_s
  launch_gemm(frelb, Wt[2], fWr_b, fr,  kR, kH, kH, 1, 0, 0, 0, stream);  // f_rel @ f_W_r
  launch_gemm(hgsb,  Wt[3], fWg_b, fg,  16, kH, kH, 1, 0, 0, 0, stream);  // h_gs @ f_W_g
  launch_gemm(rtb,   rpT,   rp_b, brelf, kR, kH, 128, 1, 0, 0, 0, stream);// rel_transe @ r_proj
  k_pad_bf16<<<blk((long)kR * kH), T, 0, stream>>>(brelf, brelb, kR, kH, kR, kH);
  launch_gemm(brelb, Wt[5], bWr_b, br,  kR, kH, kH, 1, 0, 0, 0, stream);  // b_rel @ b_W_r
  launch_gemm(hgsb,  Wt[6], bWg_b, bg,  16, kH, kH, 1, 0, 0, 0, stream);  // h_gs @ b_W_g

  // ---- tanh attention + softmax ----
  k_attn<<<ML, 256, 0, stream>>>(fx, fg, fr, V_w, V_b, fv);
  k_attn<<<ML, 256, 0, stream>>>(bx, bg, br, V_w, V_b, bv);
  k_softmax<<<blk((long)kB * kR * 32), T, 0, stream>>>(fv, fAb);
  k_softmax<<<blk((long)kB * kR * 32), T, 0, stream>>>(bv, bAb);

  // ---- context einsum via batched WMMA: C[b] = A[b](32x128) @ tok[b](128x768) ----
  launch_gemm(fAb, tokT, nullptr, fC, kR, kH, kL, kB,
              (long)kR * kL, (long)kH * kL, (long)kR * kH, stream);
  launch_gemm(bAb, tokT, nullptr, bC, kR, kH, kL, kB,
              (long)kR * kL, (long)kH * kL, (long)kR * kH, stream);

  // ---- head dots (linear decomposition of Hijk) + final sigmoid ----
  k_s2<<<blk((long)kB * kR * 32), T, 0, stream>>>(fC, bC, fso_w, feo_w, bss_w, bse_w, s2);
  k_s1<<<blk((long)ML * 32), T, 0, stream>>>(embs, fsw, fx2, bow, bx2, maskf,
                                             fso_w, feo_w, bss_w, bse_w, s1);
  k_final<<<blk(4L * kB * kL * kR), T, 0, stream>>>(s1, s2, fso_b, feo_b, bss_b, bse_b,
                                                    (float*)d_out);
}
